// CausalMultiHeadAttentionRoPE_89309549953267
// MI455X (gfx1250) — compile-verified, matrix operand order checked by reading, not run on hardware
//
#include <hip/hip_runtime.h>
#include <hip/hip_bf16.h>
#include <math.h>
#include <stdint.h>

typedef __bf16 bf16;
typedef __bf16 v16bf __attribute__((ext_vector_type(16)));
typedef float  v8f   __attribute__((ext_vector_type(8)));
typedef __bf16 bf16x4 __attribute__((ext_vector_type(4)));

#define BATCH  2
#define CTX    2048
#define DMODEL 1024
#define NHEADS 16
#define HDIM   64
#define MROWS  (BATCH*CTX)   // 4096

// ------------------------------------------------------------------
// CDNA5 async global->LDS copy (bypasses VGPRs, tracked by ASYNCcnt).
// LDS address: low 32 bits of the generic address are the LDS offset
// (ISA 10.2 aperture mapping).
// ------------------------------------------------------------------
__device__ __forceinline__ void async_copy_b128(const bf16* lds_dst, const bf16* gsrc) {
  unsigned l = (unsigned)(uintptr_t)lds_dst;
  asm volatile("global_load_async_to_lds_b128 %0, %1, off"
               :: "v"(l), "v"(gsrc) : "memory");
}
__device__ __forceinline__ void wait_async_all() {
  asm volatile("s_wait_asynccnt 0" ::: "memory");
}

union FragBF { v16bf v; bf16x4 q4[4]; };

// ------------------------------------------------------------------
// fp32 -> bf16 convert (x): vectorized
// ------------------------------------------------------------------
__global__ void cvt_kernel(const float* __restrict__ in, bf16* __restrict__ out, int n) {
  int i = (blockIdx.x * blockDim.x + threadIdx.x) * 4;
  if (i < n) {
    float4 v = *(const float4*)(in + i);
    bf16x4 o; o.x = (bf16)v.x; o.y = (bf16)v.y; o.z = (bf16)v.z; o.w = (bf16)v.w;
    *(bf16x4*)(out + i) = o;
  }
}

// ------------------------------------------------------------------
// fp32 -> bf16 convert + transpose (weights, 1024x1024):
// out[n][k] = in[k][n]; one-time 4MB job, cost negligible vs GEMMs.
// ------------------------------------------------------------------
__global__ void cvtT_kernel(const float* __restrict__ in, bf16* __restrict__ out) {
  int idx = blockIdx.x * blockDim.x + threadIdx.x;     // over 1M
  int k = idx >> 10, n = idx & 1023;
  out[(size_t)n * DMODEL + k] = (bf16)in[idx];
}

// ------------------------------------------------------------------
// WMMA bf16 GEMM: C[4096,1024] = A[4096,1024] x Bt^T  (Bt is [N][K])
// 128x128 block tile, K-step 32, double-buffered async LDS staging,
// one barrier per K-step. 8 waves (4Mx2N), wave tile 32x64, 8 WMMA/step.
// mode 0/1: RoPE epilogue (Q/K) -> head-major bf16 [b][h][pos][d]
// mode 2  : plain epilogue  (V) -> transposed bf16 [b][h][d][pos]
// mode 3  : + bias          (O) -> row-major fp32 d_out
// ------------------------------------------------------------------
#define BM 128
#define BN 128
#define BK 32
#define LDT 40   // 80B pitch: 16B-aligned async b128 chunks, conflict-free b64 gathers

__global__ __launch_bounds__(256) void gemm_bf16(
    const bf16* __restrict__ A,
    const bf16* __restrict__ Btbase, size_t bStride,
    bf16* __restrict__ dstBF, size_t dStride,
    float* __restrict__ dstF32, const float* __restrict__ bo,
    int mode_base)
{
  int z = blockIdx.z;
  int mode = mode_base + z;
  const bf16* Bt = Btbase + (size_t)z * bStride;
  bf16* dst = dstBF + (size_t)z * dStride;

  __shared__ bf16 sA[2][BM * LDT];
  __shared__ bf16 sB[2][BN * LDT];   // Bt tile: [n][k]

  int t = threadIdx.x;
  int wid = t >> 5, lane = t & 31, r = lane & 15, hl = lane >> 4;
  int wm = wid >> 1, wn = wid & 1;

  int m0 = blockIdx.y * BM;
  int n0 = blockIdx.x * BN;

  // staging coordinates: 512 16B-chunks per tile, 2 per thread
  int srow0 = t >> 2,            skc0 = t & 3;
  int srow1 = (t + 256) >> 2,    skc1 = (t + 256) & 3;

  v8f vzero = {};
  v8f acc[2][4];
#pragma unroll
  for (int a = 0; a < 2; a++)
#pragma unroll
    for (int b = 0; b < 4; b++) acc[a][b] = vzero;

  const int NT = DMODEL / BK;   // 32

  // prologue: stage tile 0 into buffer 0
  async_copy_b128(&sA[0][srow0 * LDT + skc0 * 8], A  + (size_t)(m0 + srow0) * DMODEL + skc0 * 8);
  async_copy_b128(&sB[0][srow0 * LDT + skc0 * 8], Bt + (size_t)(n0 + srow0) * DMODEL + skc0 * 8);
  async_copy_b128(&sA[0][srow1 * LDT + skc1 * 8], A  + (size_t)(m0 + srow1) * DMODEL + skc1 * 8);
  async_copy_b128(&sB[0][srow1 * LDT + skc1 * 8], Bt + (size_t)(n0 + srow1) * DMODEL + skc1 * 8);

  for (int kt = 0; kt < NT; kt++) {
    int buf = kt & 1;
    wait_async_all();        // tile kt resident (this wave's asyncs)
    __syncthreads();         // all waves' tile-kt data in LDS; old buf fully read

    if (kt + 1 < NT) {       // overlap: stage tile kt+1 into the other buffer
      int k1 = (kt + 1) * BK;
      async_copy_b128(&sA[buf ^ 1][srow0 * LDT + skc0 * 8], A  + (size_t)(m0 + srow0) * DMODEL + k1 + skc0 * 8);
      async_copy_b128(&sB[buf ^ 1][srow0 * LDT + skc0 * 8], Bt + (size_t)(n0 + srow0) * DMODEL + k1 + skc0 * 8);
      async_copy_b128(&sA[buf ^ 1][srow1 * LDT + skc1 * 8], A  + (size_t)(m0 + srow1) * DMODEL + k1 + skc1 * 8);
      async_copy_b128(&sB[buf ^ 1][srow1 * LDT + skc1 * 8], Bt + (size_t)(n0 + srow1) * DMODEL + k1 + skc1 * 8);
    }

    FragBF fa[2], fb[4];
#pragma unroll
    for (int mt = 0; mt < 2; mt++) {
      const bf16* p = &sA[buf][(wm * 32 + mt * 16 + r) * LDT + hl * 8];
      fa[mt].q4[0] = *(const bf16x4*)(p);
      fa[mt].q4[1] = *(const bf16x4*)(p + 4);
      fa[mt].q4[2] = *(const bf16x4*)(p + 16);
      fa[mt].q4[3] = *(const bf16x4*)(p + 20);
    }
#pragma unroll
    for (int nt = 0; nt < 4; nt++) {
      const bf16* p = &sB[buf][(wn * 64 + nt * 16 + r) * LDT + hl * 8];
      fb[nt].q4[0] = *(const bf16x4*)(p);
      fb[nt].q4[1] = *(const bf16x4*)(p + 4);
      fb[nt].q4[2] = *(const bf16x4*)(p + 16);
      fb[nt].q4[3] = *(const bf16x4*)(p + 20);
    }
#pragma unroll
    for (int mt = 0; mt < 2; mt++)
#pragma unroll
      for (int nt = 0; nt < 4; nt++)
        acc[mt][nt] = __builtin_amdgcn_wmma_f32_16x16x32_bf16(
            false, fa[mt].v, false, fb[nt].v, (short)0, acc[mt][nt], false, false);
  }

  // ---- epilogue
#pragma unroll
  for (int mt = 0; mt < 2; mt++) {
#pragma unroll
    for (int j = 0; j < 8; j++) {
      int grow = m0 + wm * 32 + mt * 16 + j + 8 * hl;
      int b = grow >> 11;
      int pos = grow & 2047;
#pragma unroll
      for (int nt = 0; nt < 4; nt++) {
        int col = n0 + wn * 64 + nt * 16 + r;
        float val = acc[mt][nt][j];
        if (mode < 3) {
          int h = col >> 6, d = col & 63;
          if (mode < 2) {
            // RoPE: wave's 64 N-cols align to one head -> d+/-32 partner is nt^2
            float partner = acc[mt][nt ^ 2][j];
            int dm = d & 31;
            float theta = __expf(-(float)dm * 0.5756462732485114f); // 10000^(-dm/16)
            float s = __sinf((float)pos * theta);
            float sgn = (d < 32) ? -1.f : 1.f;
            float outv = (val + sgn * partner) * s;  // reference uses sin for both terms
            dst[((size_t)(b * NHEADS + h) * CTX + pos) * HDIM + d] = (bf16)outv;
          } else {
            // V stored transposed [b][h][d][pos] so flash can async-copy V^T
            dst[((size_t)(b * NHEADS + h) * HDIM + d) * CTX + pos] = (bf16)val;
          }
        } else {
          dstF32[(size_t)grow * DMODEL + col] = val + bo[col];
        }
      }
    }
  }
}

// ------------------------------------------------------------------
// Flash attention: 128 q-rows/WG, 8 waves x 16 rows, 64-key blocks,
// async double-buffered K / V^T staging, one barrier per block.
// ------------------------------------------------------------------
#define QTILE 128
#define KBLK  64
#define LDK   72   // 144B pitch: 16B-aligned async chunks, conflict-free gathers
#define LDP   68

__global__ __launch_bounds__(256) void flash_kernel(
    const bf16* __restrict__ Qh, const bf16* __restrict__ Kh,
    const bf16* __restrict__ Vt, bf16* __restrict__ Y)
{
  __shared__ bf16 sK [2][KBLK * LDK];      // [key][d]
  __shared__ bf16 sVt[2][HDIM * LDK];      // [d][key]
  __shared__ bf16 sP [8 * 16 * LDP];       // per-wave P round-trip

  int t = threadIdx.x, wid = t >> 5, lane = t & 31, r = lane & 15, hl = lane >> 4;
  int q0 = blockIdx.x * QTILE;
  int h = blockIdx.y, b = blockIdx.z;
  size_t bh = (size_t)(b * NHEADS + h) * CTX * HDIM;   // same size for K and V^T

  // staging coords: 512 16B chunks per tensor, 2 per thread
  int srow0 = t >> 3,         sc0 = t & 7;
  int srow1 = (t + 256) >> 3, sc1 = (t + 256) & 7;

  // Q fragments for this wave's 16 rows, straight from global (L2-resident)
  FragBF fq[2];
  {
    const bf16* qb = Qh + bh + (size_t)(q0 + wid * 16 + r) * HDIM;
#pragma unroll
    for (int c = 0; c < 2; c++) {
      const bf16* p = qb + c * 32 + hl * 8;
      fq[c].q4[0] = *(const bf16x4*)(p);
      fq[c].q4[1] = *(const bf16x4*)(p + 4);
      fq[c].q4[2] = *(const bf16x4*)(p + 16);
      fq[c].q4[3] = *(const bf16x4*)(p + 20);
    }
  }

  v8f vzero = {};
  v8f o[4];
  float mrow[8], lrow[8];
#pragma unroll
  for (int i = 0; i < 4; i++) o[i] = vzero;
#pragma unroll
  for (int j = 0; j < 8; j++) { mrow[j] = -INFINITY; lrow[j] = 0.f; }

  int lastq = q0 + wid * 16 + 15;
  int nblk = (q0 + QTILE) / KBLK;

  // prologue: stage block 0 into buffer 0
  async_copy_b128(&sK[0][srow0 * LDK + sc0 * 8],  Kh + bh + (size_t)srow0 * HDIM + sc0 * 8);
  async_copy_b128(&sVt[0][srow0 * LDK + sc0 * 8], Vt + bh + (size_t)srow0 * CTX + sc0 * 8);
  async_copy_b128(&sK[0][srow1 * LDK + sc1 * 8],  Kh + bh + (size_t)srow1 * HDIM + sc1 * 8);
  async_copy_b128(&sVt[0][srow1 * LDK + sc1 * 8], Vt + bh + (size_t)srow1 * CTX + sc1 * 8);

  for (int jb = 0; jb < nblk; jb++) {
    int buf = jb & 1;
    wait_async_all();
    __syncthreads();

    if (jb + 1 < nblk) {     // overlap next block's staging with compute
      int kb = (jb + 1) * KBLK;
      async_copy_b128(&sK[buf ^ 1][srow0 * LDK + sc0 * 8],  Kh + bh + (size_t)(kb + srow0) * HDIM + sc0 * 8);
      async_copy_b128(&sVt[buf ^ 1][srow0 * LDK + sc0 * 8], Vt + bh + (size_t)srow0 * CTX + kb + sc0 * 8);
      async_copy_b128(&sK[buf ^ 1][srow1 * LDK + sc1 * 8],  Kh + bh + (size_t)(kb + srow1) * HDIM + sc1 * 8);
      async_copy_b128(&sVt[buf ^ 1][srow1 * LDK + sc1 * 8], Vt + bh + (size_t)srow1 * CTX + kb + sc1 * 8);
    }

    if (jb * KBLK <= lastq) {       // wave-uniform: skip blocks above diagonal
      // ---- S = Q K^T  (16 q x 64 keys)
      v8f s[4];
#pragma unroll
      for (int nt = 0; nt < 4; nt++) {
        FragBF fk0, fk1;
        const bf16* p = &sK[buf][(nt * 16 + r) * LDK + hl * 8];
        fk0.q4[0] = *(const bf16x4*)(p);      fk0.q4[1] = *(const bf16x4*)(p + 4);
        fk0.q4[2] = *(const bf16x4*)(p + 16); fk0.q4[3] = *(const bf16x4*)(p + 20);
        p += 32;
        fk1.q4[0] = *(const bf16x4*)(p);      fk1.q4[1] = *(const bf16x4*)(p + 4);
        fk1.q4[2] = *(const bf16x4*)(p + 16); fk1.q4[3] = *(const bf16x4*)(p + 20);
        v8f zacc = vzero;
        zacc = __builtin_amdgcn_wmma_f32_16x16x32_bf16(false, fq[0].v, false, fk0.v, (short)0, zacc, false, false);
        zacc = __builtin_amdgcn_wmma_f32_16x16x32_bf16(false, fq[1].v, false, fk1.v, (short)0, zacc, false, false);
        s[nt] = zacc;
      }
      // ---- causal mask + scale + online softmax
      float pvals[4][8];
#pragma unroll
      for (int j = 0; j < 8; j++) {
        int q = q0 + wid * 16 + j + 8 * hl;
        float mx = -INFINITY;
#pragma unroll
        for (int nt = 0; nt < 4; nt++) {
          int kg = jb * KBLK + nt * 16 + r;
          float x = (kg <= q) ? s[nt][j] * 0.125f : -INFINITY;  // 1/sqrt(64)
          pvals[nt][j] = x;
          mx = fmaxf(mx, x);
        }
        mx = fmaxf(mx, __shfl_xor(mx, 1, 32));
        mx = fmaxf(mx, __shfl_xor(mx, 2, 32));
        mx = fmaxf(mx, __shfl_xor(mx, 4, 32));
        mx = fmaxf(mx, __shfl_xor(mx, 8, 32));
        float mnew = fmaxf(mrow[j], mx);
        float corr = __expf(mrow[j] - mnew);
        float rs = 0.f;
#pragma unroll
        for (int nt = 0; nt < 4; nt++) {
          float pv = __expf(pvals[nt][j] - mnew);
          pvals[nt][j] = pv;
          rs += pv;
        }
        rs += __shfl_xor(rs, 1, 32);
        rs += __shfl_xor(rs, 2, 32);
        rs += __shfl_xor(rs, 4, 32);
        rs += __shfl_xor(rs, 8, 32);
        lrow[j] = lrow[j] * corr + rs;
        mrow[j] = mnew;
#pragma unroll
        for (int nt = 0; nt < 4; nt++) o[nt][j] *= corr;
        // C-layout -> A-fragment reshape via per-wave LDS (same-wave LDS in-order)
#pragma unroll
        for (int nt = 0; nt < 4; nt++)
          sP[(wid * 16 + j + 8 * hl) * LDP + nt * 16 + r] = (bf16)pvals[nt][j];
      }
      // ---- O += P V
      FragBF fp[2];
#pragma unroll
      for (int c = 0; c < 2; c++) {
        const bf16* p = &sP[(wid * 16 + r) * LDP + c * 32 + hl * 8];
        fp[c].q4[0] = *(const bf16x4*)(p);      fp[c].q4[1] = *(const bf16x4*)(p + 4);
        fp[c].q4[2] = *(const bf16x4*)(p + 16); fp[c].q4[3] = *(const bf16x4*)(p + 20);
      }
#pragma unroll
      for (int nt = 0; nt < 4; nt++) {
        FragBF fv0, fv1;
        const bf16* p = &sVt[buf][(nt * 16 + r) * LDK + hl * 8];
        fv0.q4[0] = *(const bf16x4*)(p);      fv0.q4[1] = *(const bf16x4*)(p + 4);
        fv0.q4[2] = *(const bf16x4*)(p + 16); fv0.q4[3] = *(const bf16x4*)(p + 20);
        p += 32;
        fv1.q4[0] = *(const bf16x4*)(p);      fv1.q4[1] = *(const bf16x4*)(p + 4);
        fv1.q4[2] = *(const bf16x4*)(p + 16); fv1.q4[3] = *(const bf16x4*)(p + 20);
        o[nt] = __builtin_amdgcn_wmma_f32_16x16x32_bf16(false, fp[0].v, false, fv0.v, (short)0, o[nt], false, false);
        o[nt] = __builtin_amdgcn_wmma_f32_16x16x32_bf16(false, fp[1].v, false, fv1.v, (short)0, o[nt], false, false);
      }
    }
  }

  // ---- normalize + write y (row-major [tok][h*64+d], bf16)
#pragma unroll
  for (int nt = 0; nt < 4; nt++)
#pragma unroll
    for (int j = 0; j < 8; j++) {
      int q = q0 + wid * 16 + j + 8 * hl;
      float outv = o[nt][j] / lrow[j];
      Y[(size_t)(b * CTX + q) * DMODEL + h * HDIM + nt * 16 + r] = (bf16)outv;
    }
}

// ------------------------------------------------------------------
extern "C" void kernel_launch(void* const* d_in, const int* in_sizes, int n_in,
                              void* d_out, int out_size, void* d_ws, size_t ws_size,
                              hipStream_t stream) {
  const float* x  = (const float*)d_in[0];
  const float* Wq = (const float*)d_in[1];
  const float* Wk = (const float*)d_in[2];
  const float* Wv = (const float*)d_in[3];
  const float* Wo = (const float*)d_in[4];
  const float* bo = (const float*)d_in[5];
  float* out = (float*)d_out;

  char* ws = (char*)d_ws;
  size_t off = 0;
  bf16* xb   = (bf16*)(ws + off); off += (size_t)MROWS * DMODEL * 2;      // 8 MB
  bf16* wqkv = (bf16*)(ws + off); off += 3 * (size_t)DMODEL * DMODEL * 2; // 6 MB (transposed)
  bf16* wob  = (bf16*)(ws + off); off += (size_t)DMODEL * DMODEL * 2;     // 2 MB (transposed)
  bf16* qkvh = (bf16*)(ws + off); off += 3 * (size_t)MROWS * DMODEL * 2;  // 24 MB
  bf16* yb   = (bf16*)(ws + off); off += (size_t)MROWS * DMODEL * 2;      // 8 MB

  const int nx = MROWS * DMODEL;      // 4M
  const int nw = DMODEL * DMODEL;     // 1M
  cvt_kernel <<<nx / 4 / 256, 256, 0, stream>>>(x, xb, nx);
  cvtT_kernel<<<nw / 256, 256, 0, stream>>>(Wq, wqkv);
  cvtT_kernel<<<nw / 256, 256, 0, stream>>>(Wk, wqkv + (size_t)nw);
  cvtT_kernel<<<nw / 256, 256, 0, stream>>>(Wv, wqkv + 2 * (size_t)nw);
  cvtT_kernel<<<nw / 256, 256, 0, stream>>>(Wo, wob);

  size_t headSz = (size_t)MROWS * DMODEL;  // elems per Q/K/V tensor

  // Q, K, V projections (+RoPE for Q,K; V written transposed) in one launch
  gemm_bf16<<<dim3(DMODEL / BN, MROWS / BM, 3), 256, 0, stream>>>(
      xb, wqkv, (size_t)nw, qkvh, headSz, nullptr, nullptr, 0);

  flash_kernel<<<dim3(CTX / QTILE, NHEADS, BATCH), 256, 0, stream>>>(
      qkvh, qkvh + headSz, qkvh + 2 * headSz, yb);

  // out = y @ Wo + bo  (fp32)
  gemm_bf16<<<dim3(DMODEL / BN, MROWS / BM, 1), 256, 0, stream>>>(
      yb, wob, 0, nullptr, 0, out, bo, 3);
}
